// SampledSoftmaxPredictionHead_34162169873007
// MI455X (gfx1250) — matrix operand order repaired and checked by reading, not compiled, not run on hardware
//
#include <hip/hip_runtime.h>

typedef __attribute__((ext_vector_type(16))) __bf16 v16bf;
typedef __attribute__((ext_vector_type(8)))  float  v8f;

#define DDIM 128
#define ROWS_PER_BLOCK 128
#define LDS_STRIDE 136          // padded bf16 row stride (272 B) -> conflict-free A-frag loads
#define MAX_NNEG 2048

// round-to-nearest-even float -> bf16 bits
static __device__ __forceinline__ unsigned short f2bf(float f) {
    unsigned int u = __float_as_uint(f);
    unsigned int r = (u + 0x7FFFu + ((u >> 16) & 1u)) >> 16;
    return (unsigned short)r;
}

struct Frag32B { uint4 lo, hi; };   // 32 bytes = 16 bf16

static __device__ __forceinline__ v16bf load_frag(const unsigned short* p) {
    Frag32B f;
    const uint4* q = reinterpret_cast<const uint4*>(p);
    f.lo = q[0]; f.hi = q[1];
    return __builtin_bit_cast(v16bf, f);
}

// one 16x16 tile of logits folded into the per-row online logsumexp state
static __device__ __forceinline__ void lse_update(
    const v8f& c, float temp, float lq, int sc, const int* yv,
    float* mx, float* sm)
{
    #pragma unroll
    for (int r = 0; r < 8; ++r) {
        float v = c[r] * temp;
        v = (yv[r] == sc) ? -1.0e9f : v;
        v -= lq;
        const float d  = v - mx[r];
        const bool  gt = d > 0.f;
        const float e  = __expf(gt ? -d : d);      // exp(-|d|): 1 exp per element
        sm[r] = gt ? __fmaf_rn(sm[r], e, 1.0f) : (sm[r] + e);
        mx[r] = gt ? v : mx[r];
    }
}

// ---------------- Kernel 1: normalize negatives -> bf16, log(samp_p+eps) ----------------
__global__ __launch_bounds__(256) void ssph_prep_neg(
    const float* __restrict__ embed, const float* __restrict__ probs,
    const int* __restrict__ sidx, unsigned short* __restrict__ negT,
    float* __restrict__ lgp, int nneg)
{
    const int wave = threadIdx.x >> 5;
    const int lane = threadIdx.x & 31;
    const int row  = blockIdx.x * 8 + wave;
    if (row >= nneg) return;
    const int e = sidx[row];
    const float4 v = reinterpret_cast<const float4*>(embed + (size_t)e * DDIM)[lane];
    float ss = v.x*v.x + v.y*v.y + v.z*v.z + v.w*v.w;
    #pragma unroll
    for (int off = 16; off > 0; off >>= 1) ss += __shfl_xor(ss, off);
    const float inv = 1.0f / fmaxf(sqrtf(ss), 1e-12f);
    uint2 pk;
    pk.x = (unsigned int)f2bf(v.x*inv) | ((unsigned int)f2bf(v.y*inv) << 16);
    pk.y = (unsigned int)f2bf(v.z*inv) | ((unsigned int)f2bf(v.w*inv) << 16);
    *reinterpret_cast<uint2*>(negT + (size_t)row * DDIM + lane * 4) = pk;
    if (lane == 0) lgp[row] = logf(probs[e] + 1e-10f);
}

// ---------------- Kernel 2: WMMA GEMM + fused online logsumexp ----------------
__global__ __launch_bounds__(256) void ssph_main(
    const float* __restrict__ hidden, const int* __restrict__ y,
    const float* __restrict__ embed, const float* __restrict__ probs,
    const float* __restrict__ logt,
    const int* __restrict__ sidx_g, const unsigned short* __restrict__ negT,
    const float* __restrict__ lgp_g, float2* __restrict__ partials,
    int Nrows, int nneg)
{
    __shared__ unsigned short hnLds[ROWS_PER_BLOCK * LDS_STRIDE];
    __shared__ float posAdjLds[ROWS_PER_BLOCK];
    __shared__ int   yLds[ROWS_PER_BLOCK];
    __shared__ int   sIdxLds[MAX_NNEG];
    __shared__ float lgpLds[MAX_NNEG];
    __shared__ float redL[16], redW[16];

    const int t = threadIdx.x;
    const float temp = __expf(fminf(logt[0], 4.6f));
    const int blockRow0 = blockIdx.x * ROWS_PER_BLOCK;

    for (int i = t; i < nneg; i += 256) { sIdxLds[i] = sidx_g[i]; lgpLds[i] = lgp_g[i]; }

    // ---- staging: normalize hidden rows -> bf16 LDS; compute pos_adj per row ----
    {
        const int row  = t >> 1;          // 2 threads per row
        const int half = t & 1;
        const int grow = blockRow0 + row;
        const bool vr  = grow < Nrows;
        const int srow = vr ? grow : 0;
        const int yrow = vr ? y[grow] : 0;
        const float4* hp = reinterpret_cast<const float4*>(hidden + (size_t)srow * DDIM + half * 64);
        const float4* ep = reinterpret_cast<const float4*>(embed  + (size_t)yrow * DDIM + half * 64);
        float ssh = 0.f, sse = 0.f, dot = 0.f;
        #pragma unroll
        for (int i = 0; i < 16; ++i) {
            float4 hv = hp[i], ev = ep[i];
            ssh += hv.x*hv.x + hv.y*hv.y + hv.z*hv.z + hv.w*hv.w;
            sse += ev.x*ev.x + ev.y*ev.y + ev.z*ev.z + ev.w*ev.w;
            dot += hv.x*ev.x + hv.y*ev.y + hv.z*ev.z + hv.w*ev.w;
        }
        ssh += __shfl_xor(ssh, 1);
        sse += __shfl_xor(sse, 1);
        dot += __shfl_xor(dot, 1);
        const float inv_h = 1.0f / fmaxf(sqrtf(ssh), 1e-12f);
        const float inv_e = 1.0f / fmaxf(sqrtf(sse), 1e-12f);
        if (half == 0) {
            posAdjLds[row] = dot * inv_h * inv_e * temp - logf(probs[yrow] + 1e-10f);
            yLds[row] = yrow;
        }
        #pragma unroll
        for (int i = 0; i < 16; ++i) {
            float4 hv = hp[i];
            uint2 pk;
            pk.x = (unsigned int)f2bf(hv.x*inv_h) | ((unsigned int)f2bf(hv.y*inv_h) << 16);
            pk.y = (unsigned int)f2bf(hv.z*inv_h) | ((unsigned int)f2bf(hv.w*inv_h) << 16);
            *reinterpret_cast<uint2*>(&hnLds[row * LDS_STRIDE + half * 64 + i * 4]) = pk;
        }
    }
    __syncthreads();

    // ---- WMMA phase: each wave owns rows [16w, 16w+16) ----
    const int w    = t >> 5;
    const int lane = t & 31;
    const int n16  = lane & 15;     // column / row-within-strip
    const int g    = lane >> 4;     // lane half

    // A fragments (held for the whole sweep): dwords 0-7: K=16g..16g+15
    v16bf afrag[4];
    #pragma unroll
    for (int kc = 0; kc < 4; ++kc) {
        const unsigned short* ab = &hnLds[(16*w + n16) * LDS_STRIDE + kc*32 + 16*g];
        Frag32B f;
        f.lo = *reinterpret_cast<const uint4*>(ab);
        f.hi = *reinterpret_cast<const uint4*>(ab + 8);
        afrag[kc] = __builtin_bit_cast(v16bf, f);
    }
    int yv[8];
    #pragma unroll
    for (int r = 0; r < 8; ++r) yv[r] = yLds[16*w + r + 8*g];

    float mx[8], sm[8];
    #pragma unroll
    for (int r = 0; r < 8; ++r) { mx[r] = -3.0e38f; sm[r] = 0.f; }

    const int ntiles = nneg >> 4;

    // 2-way unrolled sweep: two independent WMMA accumulator chains in flight
    int jt = 0;
    for (; jt + 1 < ntiles; jt += 2) {
        const int col0 = (jt << 4) + n16;
        const int col1 = col0 + 16;
        const unsigned short* bb0 = negT + (size_t)col0 * DDIM + 16*g;
        const unsigned short* bb1 = negT + (size_t)col1 * DDIM + 16*g;

        v16bf b00 = load_frag(bb0);
        v16bf b01 = load_frag(bb0 + 32);
        v16bf b02 = load_frag(bb0 + 64);
        v16bf b03 = load_frag(bb0 + 96);
        v16bf b10 = load_frag(bb1);
        v16bf b11 = load_frag(bb1 + 32);
        v16bf b12 = load_frag(bb1 + 64);
        v16bf b13 = load_frag(bb1 + 96);

        v8f c0 = {0.f,0.f,0.f,0.f,0.f,0.f,0.f,0.f};
        v8f c1 = {0.f,0.f,0.f,0.f,0.f,0.f,0.f,0.f};
        // interleave the two independent chains
        c0 = __builtin_amdgcn_wmma_f32_16x16x32_bf16(false, afrag[0], false, b00, (short)0, c0, false, false);
        c1 = __builtin_amdgcn_wmma_f32_16x16x32_bf16(false, afrag[0], false, b10, (short)0, c1, false, false);
        c0 = __builtin_amdgcn_wmma_f32_16x16x32_bf16(false, afrag[1], false, b01, (short)0, c0, false, false);
        c1 = __builtin_amdgcn_wmma_f32_16x16x32_bf16(false, afrag[1], false, b11, (short)0, c1, false, false);
        c0 = __builtin_amdgcn_wmma_f32_16x16x32_bf16(false, afrag[2], false, b02, (short)0, c0, false, false);
        c1 = __builtin_amdgcn_wmma_f32_16x16x32_bf16(false, afrag[2], false, b12, (short)0, c1, false, false);
        c0 = __builtin_amdgcn_wmma_f32_16x16x32_bf16(false, afrag[3], false, b03, (short)0, c0, false, false);
        c1 = __builtin_amdgcn_wmma_f32_16x16x32_bf16(false, afrag[3], false, b13, (short)0, c1, false, false);

        lse_update(c0, temp, lgpLds[col0], sIdxLds[col0], yv, mx, sm);
        lse_update(c1, temp, lgpLds[col1], sIdxLds[col1], yv, mx, sm);
    }
    for (; jt < ntiles; ++jt) {   // tail (unused when ntiles is even)
        const int col = (jt << 4) + n16;
        const unsigned short* bb = negT + (size_t)col * DDIM + 16*g;
        v16bf b0 = load_frag(bb);
        v16bf b1 = load_frag(bb + 32);
        v16bf b2 = load_frag(bb + 64);
        v16bf b3 = load_frag(bb + 96);
        v8f c = {0.f,0.f,0.f,0.f,0.f,0.f,0.f,0.f};
        c = __builtin_amdgcn_wmma_f32_16x16x32_bf16(false, afrag[0], false, b0, (short)0, c, false, false);
        c = __builtin_amdgcn_wmma_f32_16x16x32_bf16(false, afrag[1], false, b1, (short)0, c, false, false);
        c = __builtin_amdgcn_wmma_f32_16x16x32_bf16(false, afrag[2], false, b2, (short)0, c, false, false);
        c = __builtin_amdgcn_wmma_f32_16x16x32_bf16(false, afrag[3], false, b3, (short)0, c, false, false);
        lse_update(c, temp, lgpLds[col], sIdxLds[col], yv, mx, sm);
    }

    // merge (mx,sm) across the 16 lanes sharing each row
    #pragma unroll
    for (int r = 0; r < 8; ++r) {
        #pragma unroll
        for (int off = 1; off < 16; off <<= 1) {
            const float om = __shfl_xor(mx[r], off);
            const float os = __shfl_xor(sm[r], off);
            const float nm = fmaxf(mx[r], om);
            sm[r] = sm[r]*__expf(mx[r]-nm) + os*__expf(om-nm);
            mx[r] = nm;
        }
    }

    float lsum = 0.f, wsum = 0.f;
    if (n16 == 0) {  // lane 0: rows r; lane 16: rows r+8
        #pragma unroll
        for (int r = 0; r < 8; ++r) {
            const int rl = 16*w + r + 8*g;
            const int grow = blockRow0 + rl;
            const float pa = posAdjLds[rl];
            const float nm = fmaxf(mx[r], pa);
            const float s  = sm[r]*__expf(mx[r]-nm) + __expf(pa-nm);
            const float loss = nm + logf(s) - pa;
            if (grow < Nrows && yLds[rl] != 0) { lsum += loss; wsum += 1.0f; }
        }
        redL[w*2 + g] = lsum;
        redW[w*2 + g] = wsum;
    }
    __syncthreads();
    if (t == 0) {
        float L = 0.f, W = 0.f;
        #pragma unroll
        for (int i = 0; i < 16; ++i) { L += redL[i]; W += redW[i]; }
        partials[blockIdx.x] = make_float2(L, W);
    }
}

// ---------------- Kernel 3: deterministic fixed-order reduction ----------------
__global__ __launch_bounds__(256) void ssph_finalize(
    const float2* __restrict__ partials, int nblocks, float* __restrict__ out)
{
    __shared__ float sL[256], sW[256];
    const int t = threadIdx.x;
    float L = 0.f, W = 0.f;
    for (int i = t; i < nblocks; i += 256) { L += partials[i].x; W += partials[i].y; }
    sL[t] = L; sW[t] = W;
    __syncthreads();
    #pragma unroll
    for (int s = 128; s > 0; s >>= 1) {
        if (t < s) { sL[t] += sL[t + s]; sW[t] += sW[t + s]; }
        __syncthreads();
    }
    if (t == 0) out[0] = sL[0] / fmaxf(sW[0], 1.0f);
}

extern "C" void kernel_launch(void* const* d_in, const int* in_sizes, int n_in,
                              void* d_out, int out_size, void* d_ws, size_t ws_size,
                              hipStream_t stream) {
    const float* hidden = (const float*)d_in[0];
    const int*   y      = (const int*)d_in[1];
    const float* embed  = (const float*)d_in[2];
    const float* probs  = (const float*)d_in[3];
    const int*   sidx   = (const int*)d_in[4];
    const float* logt   = (const float*)d_in[5];
    float* out = (float*)d_out;

    const int Nrows = in_sizes[0] / DDIM;       // 65536
    const int nneg  = in_sizes[4];              // 2048

    char* ws = (char*)d_ws;
    size_t off = 0;
    unsigned short* negT = (unsigned short*)(ws + off);
    off += (size_t)nneg * DDIM * sizeof(unsigned short);
    off = (off + 255) & ~(size_t)255;
    float* lgp = (float*)(ws + off);
    off += (size_t)nneg * sizeof(float);
    off = (off + 255) & ~(size_t)255;
    float2* partials = (float2*)(ws + off);

    const int nbPrep = (nneg + 7) / 8;
    const int nbMain = (Nrows + ROWS_PER_BLOCK - 1) / ROWS_PER_BLOCK;

    ssph_prep_neg<<<nbPrep, 256, 0, stream>>>(embed, probs, sidx, negT, lgp, nneg);
    ssph_main<<<nbMain, 256, 0, stream>>>(hidden, y, embed, probs, logt,
                                          sidx, negT, lgp, partials, Nrows, nneg);
    ssph_finalize<<<1, 256, 0, stream>>>(partials, nbMain, out);
}
